// BEVFormerLayer_32504312496505
// MI455X (gfx1250) — matrix operand order repaired and checked by reading, not compiled, not run on hardware
//
#include <hip/hip_runtime.h>
#include <hip/hip_bf16.h>
#include <stdint.h>
#include <stddef.h>

// ---------------------------------------------------------------------------
// BEVFormer layer forward for MI455X (gfx1250, wave32, WMMA).
// Dense linears -> v_wmma_f32_16x16x32_bf16, f32 accumulate.
// Fragments are loaded as 2x b128 per lane (A row-major, B pre-transposed),
// matching the ISA 7.12.2 16-bit operand layouts with zero repacking VALU.
// ---------------------------------------------------------------------------

typedef __attribute__((ext_vector_type(16))) __bf16          v16bf;
typedef __attribute__((ext_vector_type(16))) unsigned short  v16us;
typedef __attribute__((ext_vector_type(8)))  float           v8f;

static constexpr int kNQ    = 5440;   // num queries == NUM_VALUE
static constexpr int kE     = 256;
static constexpr int kHD    = 32;
static constexpr int kCams  = 6;

// level tables: SHAPES = (64,64),(32,32),(16,16),(8,8)
__device__ __constant__ int   c_lvlW[4] = {64, 32, 16, 8};
__device__ __constant__ int   c_lvlH[4] = {64, 32, 16, 8};
__device__ __constant__ int   c_lvlS[4] = {0, 4096, 5120, 5376};
__device__ __constant__ float c_invW[4] = {1.f/64.f, 1.f/32.f, 1.f/16.f, 1.f/8.f};
__device__ __constant__ float c_invH[4] = {1.f/64.f, 1.f/32.f, 1.f/16.f, 1.f/8.f};

// ---------------------------------------------------------------------------
// helpers
// ---------------------------------------------------------------------------
__device__ __forceinline__ unsigned short f32_to_bf16_rne(float f) {
  unsigned int u = __float_as_uint(f);
  unsigned int lsb = (u >> 16) & 1u;
  u += 0x7FFFu + lsb;                 // round-to-nearest-even
  return (unsigned short)(u >> 16);
}

// two 16B chunks -> one 32B fragment register block
__device__ __forceinline__ v16bf load_frag_2x16B(const unsigned short* p0,
                                                 const unsigned short* p1) {
  struct TwoQ { uint4 a, b; } t;
  t.a = *reinterpret_cast<const uint4*>(p0);
  t.b = *reinterpret_cast<const uint4*>(p1);
  return __builtin_bit_cast(v16bf, t);
}

// A fragment: 16x32 bf16 tile at (row0, k0) of row-major A (ld = K).
// lane = half*16 + m: holds rows m, K-runs [k0+8h .. +7] and [k0+16+8h .. +7].
__device__ __forceinline__ v16bf load_a_frag(const unsigned short* __restrict__ A,
                                             int lda, int row0, int k0,
                                             int m, int half) {
  const unsigned short* p = A + (size_t)(row0 + m) * lda + k0 + 8 * half;
  return load_frag_2x16B(p, p + 16);
}

// B fragment: 32x16 bf16 tile, B stored TRANSPOSED (N x K row-major, ld = K).
// lane = half*16 + n: holds column n, K-runs [k0+8h .. +7] and [k0+16+8h .. +7].
__device__ __forceinline__ v16bf load_b_fragT(const unsigned short* __restrict__ BT,
                                              int ldbt, int k0, int col0,
                                              int n, int half) {
  const unsigned short* p = BT + (size_t)(col0 + n) * ldbt + k0 + 8 * half;
  return load_frag_2x16B(p, p + 16);
}

// C/D 16x16 f32: VGPR r -> row (row0 + half*8 + r), col (col0 + lane%16)
__device__ __forceinline__ void store_d_frag(float* __restrict__ C, int ldc,
                                             int row0, int col0, int n,
                                             int half, v8f d,
                                             const float* __restrict__ bias) {
  int mo = half * 8;
  float bv = bias[col0 + n];
#pragma unroll
  for (int r = 0; r < 8; ++r)
    C[(size_t)(row0 + mo + r) * ldc + col0 + n] = d[r] + bv;
}

#define WMMA_BF16(A_, B_, C_)                                                  \
  __builtin_amdgcn_wmma_f32_16x16x32_bf16(false, (A_), false, (B_), (short)0,  \
                                          (C_), false, false)

// ---------------------------------------------------------------------------
// GEMM: C[M,N] = A[M,K](bf16,row-major) * B[K,N] + bias[N], f32 out.
// B passed transposed (BT is N x K, row-major). One wave per 64x64 C tile:
// 4 A-frags x 4 B-frags -> 16 WMMAs per 32-deep K step, 16 b128 loads.
// M, N multiples of 64; K multiple of 32. Block = 256 threads (8 waves).
// ---------------------------------------------------------------------------
__global__ void __launch_bounds__(256)
gemm_bf16_wmma(const unsigned short* __restrict__ A,
               const unsigned short* __restrict__ BT,
               const float* __restrict__ bias, float* __restrict__ C,
               int M, int N, int K) {
  int gwave = (int)((blockIdx.x * blockDim.x + threadIdx.x) >> 5);
  int lane  = (int)(threadIdx.x & 31);
  int tn_cnt = N >> 6;
  int total  = (M >> 6) * tn_cnt;
  if (gwave >= total) return;
  int tm = gwave / tn_cnt, tn = gwave % tn_cnt;
  int row0 = tm << 6, col0 = tn << 6;
  int m = lane & 15, half = lane >> 4;

  v8f acc[4][4] = {{{}, {}, {}, {}}, {{}, {}, {}, {}},
                   {{}, {}, {}, {}}, {{}, {}, {}, {}}};

  for (int k0 = 0; k0 < K; k0 += 32) {
    v16bf af[4], bf[4];
#pragma unroll
    for (int i = 0; i < 4; ++i)
      af[i] = load_a_frag(A, K, row0 + 16 * i, k0, m, half);
#pragma unroll
    for (int j = 0; j < 4; ++j)
      bf[j] = load_b_fragT(BT, K, k0, col0 + 16 * j, m, half);
#pragma unroll
    for (int i = 0; i < 4; ++i)
#pragma unroll
      for (int j = 0; j < 4; ++j)
        acc[i][j] = WMMA_BF16(af[i], bf[j], acc[i][j]);
  }
#pragma unroll
  for (int i = 0; i < 4; ++i)
#pragma unroll
    for (int j = 0; j < 4; ++j)
      store_d_frag(C, N, row0 + 16 * i, col0 + 16 * j, m, half, acc[i][j],
                   bias);
}

// ---------------------------------------------------------------------------
// conversions / packing
// ---------------------------------------------------------------------------
__global__ void cvt_f32_to_bf16(const float* __restrict__ src,
                                unsigned short* __restrict__ dst, size_t n) {
  size_t i = (size_t)blockIdx.x * blockDim.x + threadIdx.x;
  if (i < n) dst[i] = f32_to_bf16_rne(src[i]);
}

// weight [K,N] f32 -> transposed bf16 [N,K]
__global__ void cvt_transpose_bf16(const float* __restrict__ src,
                                   unsigned short* __restrict__ dst, int Kdim,
                                   int Ndim) {
  int i = (int)(blockIdx.x * blockDim.x + threadIdx.x);
  if (i >= Kdim * Ndim) return;
  int k = i / Ndim, n = i - k * Ndim;
  dst[(size_t)n * Kdim + k] = f32_to_bf16_rne(src[i]);
}

// qcat[q, 0:256]=hist, [q,256:512]=query  -> bf16 [5440,512]
__global__ void build_qcat_bf16(const float* __restrict__ hist,
                                const float* __restrict__ query,
                                unsigned short* __restrict__ dst) {
  int i = (int)(blockIdx.x * blockDim.x + threadIdx.x);
  if (i >= kNQ * 2 * kE) return;
  int q = i >> 9, j = i & 511;
  float v = (j < kE) ? hist[q * kE + j] : query[q * kE + (j - kE)];
  dst[i] = f32_to_bf16_rne(v);
}

// stacked temporal value: rows 0..5439 = hist, 5440..10879 = query -> bf16
__global__ void build_valt_bf16(const float* __restrict__ hist,
                                const float* __restrict__ query,
                                unsigned short* __restrict__ dst) {
  int i = (int)(blockIdx.x * blockDim.x + threadIdx.x);
  const int half = kNQ * kE;
  if (i >= 2 * half) return;
  float v = (i < half) ? hist[i] : query[i - half];
  dst[i] = f32_to_bf16_rne(v);
}

// ---------------------------------------------------------------------------
// softmax over contiguous groups of 16
// ---------------------------------------------------------------------------
__global__ void softmax16_inplace(float* __restrict__ buf, int ngroups,
                                  int rowStride, int groupsPerRow) {
  int g = (int)(blockIdx.x * blockDim.x + threadIdx.x);
  if (g >= ngroups) return;
  int row = g / groupsPerRow, gi = g - row * groupsPerRow;
  float* p = buf + (size_t)row * rowStride + gi * 16;
  float mx = p[0];
#pragma unroll
  for (int i = 1; i < 16; ++i) mx = fmaxf(mx, p[i]);
  float e[16], s = 0.f;
#pragma unroll
  for (int i = 0; i < 16; ++i) { e[i] = expf(p[i] - mx); s += e[i]; }
  float inv = 1.f / s;
#pragma unroll
  for (int i = 0; i < 16; ++i) p[i] = e[i] * inv;
}

// ---------------------------------------------------------------------------
// bilinear sample, grid_sample(align_corners=False, zero pad) semantics
// vbase: [NUM_VALUE, 256] f32 for one batch slice; ch = head*32 + d
// ---------------------------------------------------------------------------
__device__ __forceinline__ float sample_level(const float* __restrict__ vbase,
                                              int lvl, float lx, float ly,
                                              int ch) {
  int W = c_lvlW[lvl], H = c_lvlH[lvl], st = c_lvlS[lvl];
  float x = lx * (float)W - 0.5f;
  float y = ly * (float)H - 0.5f;
  float fx = floorf(x), fy = floorf(y);
  float wx = x - fx, wy = y - fy;
  int ix0 = (int)fx, iy0 = (int)fy;
  float acc = 0.f;
#pragma unroll
  for (int dy = 0; dy < 2; ++dy) {
#pragma unroll
    for (int dx = 0; dx < 2; ++dx) {
      int ix = ix0 + dx, iy = iy0 + dy;
      if (ix >= 0 && ix < W && iy >= 0 && iy < H) {
        float w = (dx ? wx : 1.f - wx) * (dy ? wy : 1.f - wy);
        acc += w * vbase[(size_t)(st + iy * W + ix) * kE + ch];
      }
    }
  }
  return acc;
}

// ---------------------------------------------------------------------------
// temporal deformable attention: out = mean over queue of ms_deform_attn
// grid: 5440 blocks x 256 threads; wave h handles head h, lane d = channel
// ---------------------------------------------------------------------------
__global__ void __launch_bounds__(256)
temporal_sample(const float* __restrict__ val,   // [2*5440, 256]
                const float* __restrict__ off,   // [5440, 512]
                const float* __restrict__ aw,    // [5440, 256] post-softmax
                const float* __restrict__ refp,  // [5440, 4, 2]
                float* __restrict__ tout) {      // [5440, 256]
  int q = (int)blockIdx.x;
  int lane = (int)(threadIdx.x & 31);
  int h = (int)(threadIdx.x >> 5);
  int ch = h * kHD + lane;
  float acc = 0.f;
#pragma unroll
  for (int qu = 0; qu < 2; ++qu) {
    const float* vb = val + (size_t)qu * kNQ * kE;
#pragma unroll
    for (int l = 0; l < 4; ++l) {
      float rx = refp[(q * 4 + l) * 2 + 0];
      float ry = refp[(q * 4 + l) * 2 + 1];
#pragma unroll
      for (int p = 0; p < 4; ++p) {
        int oi = q * 512 + ((((h * 2 + qu) * 4 + l) * 4 + p) * 2);
        float lx = rx + off[oi]     * c_invW[l];
        float ly = ry + off[oi + 1] * c_invH[l];
        float w  = aw[q * 256 + (h * 2 + qu) * 16 + l * 4 + p];
        acc += w * sample_level(vb, l, lx, ly, ch);
      }
    }
  }
  tout[(size_t)q * kE + ch] = 0.5f * acc;
}

// ---------------------------------------------------------------------------
// spatial cross-attention sampling over cameras, mask + count, bf16 slots out
// ---------------------------------------------------------------------------
__global__ void __launch_bounds__(256)
spatial_sample(const float* __restrict__ val,    // [6*5440, 256]
               const float* __restrict__ off,    // [5440, 256]
               const float* __restrict__ aw,     // [5440, 128] post-softmax
               const float* __restrict__ refc,   // [6, 5440, 4, 2]
               const unsigned char* __restrict__ mask,  // [6, 5440, 4] bool
               unsigned short* __restrict__ slots16) {  // [5440, 256] bf16
  int q = (int)blockIdx.x;
  int lane = (int)(threadIdx.x & 31);
  int h = (int)(threadIdx.x >> 5);
  int ch = h * kHD + lane;
  float acc = 0.f, cnt = 0.f;
  for (int cam = 0; cam < kCams; ++cam) {
    const unsigned char* mp = mask + ((size_t)cam * kNQ + q) * 4;
    bool any = (mp[0] | mp[1] | mp[2] | mp[3]) != 0;
    if (any) cnt += 1.f;
    if (!any) continue;
    const float* vb = val + (size_t)cam * kNQ * kE;
    float oc = 0.f;
#pragma unroll
    for (int l = 0; l < 4; ++l) {
      float rx = refc[(((size_t)cam * kNQ + q) * 4 + l) * 2 + 0];
      float ry = refc[(((size_t)cam * kNQ + q) * 4 + l) * 2 + 1];
#pragma unroll
      for (int p = 0; p < 4; ++p) {
        int oi = q * 256 + (((h * 4 + l) * 4 + p) * 2);
        float lx = rx + off[oi]     * c_invW[l];
        float ly = ry + off[oi + 1] * c_invH[l];
        float w  = aw[q * 128 + h * 16 + l * 4 + p];
        oc += w * sample_level(vb, l, lx, ly, ch);
      }
    }
    acc += oc;
  }
  float c = cnt > 1.f ? cnt : 1.f;
  slots16[(size_t)q * kE + ch] = f32_to_bf16_rne(acc / c);
}

// ---------------------------------------------------------------------------
// y = LayerNorm(X + scaleR * R) * g + b ; optional bf16 copy of y.
// one wave per row of 256; 8 rows per block.
// ---------------------------------------------------------------------------
__global__ void __launch_bounds__(256)
add_layernorm(const float* __restrict__ X, const float* __restrict__ R,
              float scaleR, const float* __restrict__ g,
              const float* __restrict__ b, float* __restrict__ out,
              unsigned short* __restrict__ out16) {
  int row = (int)(blockIdx.x * (blockDim.x >> 5) + (threadIdx.x >> 5));
  int lane = (int)(threadIdx.x & 31);
  if (row >= kNQ) return;
  const float* xr = X + (size_t)row * kE;
  const float* rr = R + (size_t)row * kE;
  float v[8], s = 0.f;
#pragma unroll
  for (int i = 0; i < 8; ++i) {
    v[i] = xr[lane + i * 32] + scaleR * rr[lane + i * 32];
    s += v[i];
  }
#pragma unroll
  for (int o = 16; o > 0; o >>= 1) s += __shfl_xor(s, o, 32);
  float mu = s * (1.f / 256.f);
  float var = 0.f;
#pragma unroll
  for (int i = 0; i < 8; ++i) { float d = v[i] - mu; var += d * d; }
#pragma unroll
  for (int o = 16; o > 0; o >>= 1) var += __shfl_xor(var, o, 32);
  var *= (1.f / 256.f);
  float inv = rsqrtf(var + 1e-5f);
#pragma unroll
  for (int i = 0; i < 8; ++i) {
    int c = lane + i * 32;
    float y = (v[i] - mu) * inv * g[c] + b[c];
    out[(size_t)row * kE + c] = y;
    if (out16) out16[(size_t)row * kE + c] = f32_to_bf16_rne(y);
  }
}

// ---------------------------------------------------------------------------
// host launch
// ---------------------------------------------------------------------------
static inline unsigned ceil_div_u(size_t a, size_t b) {
  return (unsigned)((a + b - 1) / b);
}
static inline unsigned gemm_blocks(int M, int N) {
  int tiles = (M / 64) * (N / 64);
  return (unsigned)((tiles + 7) / 8);  // 8 waves per 256-thread block
}

extern "C" void kernel_launch(void* const* d_in, const int* in_sizes, int n_in,
                              void* d_out, int out_size, void* d_ws,
                              size_t ws_size, hipStream_t stream) {
  (void)in_sizes; (void)n_in; (void)out_size; (void)ws_size;
  // inputs (setup_inputs order; params dict flattened in insertion order)
  const float* spat_value = (const float*)d_in[1];        // [6,1,5440,256]
  const float* t_query    = (const float*)d_in[2];        // [1,5440,256]
  const float* t_val_hist = (const float*)d_in[4];        // [1,5440,256]
  const float* ref_points = (const float*)d_in[5];        // [1,5440,4,2]
  const float* ref_cam    = (const float*)d_in[7];        // [6,1,5440,4,2]
  const unsigned char* bev_mask = (const unsigned char*)d_in[8]; // [6,1,5440,4]
  const float* W_off_t  = (const float*)d_in[9];   const float* b_off_t  = (const float*)d_in[10];
  const float* W_attn_t = (const float*)d_in[11];  const float* b_attn_t = (const float*)d_in[12];
  const float* W_val_t  = (const float*)d_in[13];  const float* b_val_t  = (const float*)d_in[14];
  const float* W_out_t  = (const float*)d_in[15];  const float* b_out_t  = (const float*)d_in[16];
  const float* W_off_s  = (const float*)d_in[17];  const float* b_off_s  = (const float*)d_in[18];
  const float* W_attn_s = (const float*)d_in[19];  const float* b_attn_s = (const float*)d_in[20];
  const float* W_val_s  = (const float*)d_in[21];  const float* b_val_s  = (const float*)d_in[22];
  const float* W_out_s  = (const float*)d_in[23];  const float* b_out_s  = (const float*)d_in[24];
  const float* W_ffn    = (const float*)d_in[25];  const float* b_ffn    = (const float*)d_in[26];
  const float* ln_g     = (const float*)d_in[27];  const float* ln_b     = (const float*)d_in[28];
  float* out = (float*)d_out;

  // workspace arena (256B aligned slots -> all bf16 rows 16B aligned)
  char* ws = (char*)d_ws;
  size_t cur = 0;
  auto alloc = [&](size_t bytes) -> void* {
    void* p = ws + cur;
    cur += (bytes + 255) & ~(size_t)255;
    return p;
  };
  // bf16 transposed weights [N,K]
  unsigned short* w_off_t16  = (unsigned short*)alloc(512 * 512 * 2);
  unsigned short* w_attn_t16 = (unsigned short*)alloc(512 * 256 * 2);
  unsigned short* w_val_t16  = (unsigned short*)alloc(256 * 256 * 2);
  unsigned short* w_out_t16  = (unsigned short*)alloc(256 * 256 * 2);
  unsigned short* w_off_s16  = (unsigned short*)alloc(256 * 256 * 2);
  unsigned short* w_attn_s16 = (unsigned short*)alloc(256 * 128 * 2);
  unsigned short* w_val_s16  = (unsigned short*)alloc(256 * 256 * 2);
  unsigned short* w_out_s16  = (unsigned short*)alloc(256 * 256 * 2);
  unsigned short* w_ffn16    = (unsigned short*)alloc(256 * 256 * 2);
  // bf16 activations (row-major)
  unsigned short* qcat16  = (unsigned short*)alloc((size_t)kNQ * 512 * 2);
  unsigned short* valt16  = (unsigned short*)alloc((size_t)2 * kNQ * kE * 2);
  unsigned short* sval16  = (unsigned short*)alloc((size_t)kCams * kNQ * kE * 2);
  unsigned short* tout16  = (unsigned short*)alloc((size_t)kNQ * kE * 2);
  unsigned short* q16     = (unsigned short*)alloc((size_t)kNQ * kE * 2);
  unsigned short* slots16 = (unsigned short*)alloc((size_t)kNQ * kE * 2);
  unsigned short* an2_16  = (unsigned short*)alloc((size_t)kNQ * kE * 2);
  // f32 intermediates
  float* off_t_out = (float*)alloc((size_t)kNQ * 512 * 4);
  float* aw_t      = (float*)alloc((size_t)kNQ * 256 * 4);
  float* val_t_out = (float*)alloc((size_t)2 * kNQ * kE * 4);
  float* tout      = (float*)alloc((size_t)kNQ * kE * 4);
  float* t_lin     = (float*)alloc((size_t)kNQ * kE * 4);
  float* addnorm1  = (float*)alloc((size_t)kNQ * kE * 4);
  float* off_s_out = (float*)alloc((size_t)kNQ * kE * 4);
  float* aw_s      = (float*)alloc((size_t)kNQ * 128 * 4);
  float* sval_out  = (float*)alloc((size_t)kCams * kNQ * kE * 4);
  float* s_lin     = (float*)alloc((size_t)kNQ * kE * 4);
  float* addnorm2  = (float*)alloc((size_t)kNQ * kE * 4);
  float* ffn_out   = (float*)alloc((size_t)kNQ * kE * 4);

  const unsigned TB = 256;
  auto cvtT = [&](const float* s, unsigned short* d, int Kd, int Nd) {
    cvt_transpose_bf16<<<ceil_div_u((size_t)Kd * Nd, TB), TB, 0, stream>>>(
        s, d, Kd, Nd);
  };
  auto cvt = [&](const float* s, unsigned short* d, size_t n) {
    cvt_f32_to_bf16<<<ceil_div_u(n, TB), TB, 0, stream>>>(s, d, n);
  };

  // ---- stage 0: precision conversion / packing ----
  cvtT(W_off_t,  w_off_t16,  512, 512);
  cvtT(W_attn_t, w_attn_t16, 512, 256);
  cvtT(W_val_t,  w_val_t16,  256, 256);
  cvtT(W_out_t,  w_out_t16,  256, 256);
  cvtT(W_off_s,  w_off_s16,  256, 256);
  cvtT(W_attn_s, w_attn_s16, 256, 128);
  cvtT(W_val_s,  w_val_s16,  256, 256);
  cvtT(W_out_s,  w_out_s16,  256, 256);
  cvtT(W_ffn,    w_ffn16,    256, 256);
  build_qcat_bf16<<<ceil_div_u((size_t)kNQ * 512, TB), TB, 0, stream>>>(
      t_val_hist, t_query, qcat16);
  build_valt_bf16<<<ceil_div_u((size_t)2 * kNQ * kE, TB), TB, 0, stream>>>(
      t_val_hist, t_query, valt16);
  cvt(spat_value, sval16, (size_t)kCams * kNQ * kE);

  // ---- temporal self-attention ----
  gemm_bf16_wmma<<<gemm_blocks(kNQ, 512), TB, 0, stream>>>(
      qcat16, w_off_t16, b_off_t, off_t_out, kNQ, 512, 512);
  gemm_bf16_wmma<<<gemm_blocks(kNQ, 256), TB, 0, stream>>>(
      qcat16, w_attn_t16, b_attn_t, aw_t, kNQ, 256, 512);
  softmax16_inplace<<<ceil_div_u((size_t)kNQ * 16, TB), TB, 0, stream>>>(
      aw_t, kNQ * 16, 256, 16);
  gemm_bf16_wmma<<<gemm_blocks(2 * kNQ, 256), TB, 0, stream>>>(
      valt16, w_val_t16, b_val_t, val_t_out, 2 * kNQ, 256, 256);
  temporal_sample<<<kNQ, TB, 0, stream>>>(val_t_out, off_t_out, aw_t,
                                          ref_points, tout);
  cvt(tout, tout16, (size_t)kNQ * kE);
  gemm_bf16_wmma<<<gemm_blocks(kNQ, 256), TB, 0, stream>>>(
      tout16, w_out_t16, b_out_t, t_lin, kNQ, 256, 256);
  // addnorm1 = LN(t_lin + 2*query)   [temp_attn = lin + q; LN(temp_attn + q)]
  add_layernorm<<<ceil_div_u(kNQ, 8), TB, 0, stream>>>(
      t_lin, t_query, 2.0f, ln_g, ln_b, addnorm1, q16);

  // ---- spatial cross-attention ----
  gemm_bf16_wmma<<<gemm_blocks(kNQ, 256), TB, 0, stream>>>(
      q16, w_off_s16, b_off_s, off_s_out, kNQ, 256, 256);
  gemm_bf16_wmma<<<gemm_blocks(kNQ, 128), TB, 0, stream>>>(
      q16, w_attn_s16, b_attn_s, aw_s, kNQ, 128, 256);
  softmax16_inplace<<<ceil_div_u((size_t)kNQ * 8, TB), TB, 0, stream>>>(
      aw_s, kNQ * 8, 128, 8);
  gemm_bf16_wmma<<<gemm_blocks(kCams * kNQ, 256), TB, 0, stream>>>(
      sval16, w_val_s16, b_val_s, sval_out, kCams * kNQ, 256, 256);
  spatial_sample<<<kNQ, TB, 0, stream>>>(sval_out, off_s_out, aw_s, ref_cam,
                                         bev_mask, slots16);
  gemm_bf16_wmma<<<gemm_blocks(kNQ, 256), TB, 0, stream>>>(
      slots16, w_out_s16, b_out_s, s_lin, kNQ, 256, 256);
  // addnorm2 = LN(s_lin + 2*addnorm1)
  add_layernorm<<<ceil_div_u(kNQ, 8), TB, 0, stream>>>(
      s_lin, addnorm1, 2.0f, ln_g, ln_b, addnorm2, an2_16);

  // ---- FFN + final norm ----
  gemm_bf16_wmma<<<gemm_blocks(kNQ, 256), TB, 0, stream>>>(
      an2_16, w_ffn16, b_ffn, ffn_out, kNQ, 256, 256);
  add_layernorm<<<ceil_div_u(kNQ, 8), TB, 0, stream>>>(
      ffn_out, addnorm2, 1.0f, ln_g, ln_b, out, nullptr);
}
